// SOHO_Pre_VD_88029649699062
// MI455X (gfx1250) — matrix operand order repaired and checked by reading, not compiled, not run on hardware
//
#include <hip/hip_runtime.h>
#include <hip/hip_bf16.h>
#include <stdint.h>

// ---------------------------------------------------------------------------
// VQ nearest-codebook search for MI455X (gfx1250, wave32, WMMA).
//   N = 32768 inputs, K = 8192 codes, D = 256 dims.
//   argmin_k ( e_sq[k] - 2 * x . e_k )   (x-norm irrelevant to argmin)
//   quantize = embed[argmin]  (exact fp32 gather)
// Scores computed with v_wmma_f32_16x16x32_bf16 (f32 accumulate).
// ---------------------------------------------------------------------------

#define N_ROWS 32768
#define N_CODES 8192
#define DIM 256

typedef __bf16 v16bf __attribute__((ext_vector_type(16)));
typedef float  v8f   __attribute__((ext_vector_type(8)));

union Frag16 {            // 32 bytes = one 16-bit WMMA A/B fragment (8 VGPRs)
    v16bf v;
    uint4 q[2];
};

static __device__ __forceinline__ unsigned short f2bf_rne(float f) {
    unsigned int u = __float_as_uint(f);
    unsigned int r = (u + 0x7FFFu + ((u >> 16) & 1u)) >> 16;   // RNE
    return (unsigned short)r;
}

// ---------------------------------------------------------------------------
// Kernel 1: fp32 inputs -> bf16 workspace. 4 elements / thread.
// ---------------------------------------------------------------------------
__global__ void vq_convert_inputs(const float* __restrict__ x,
                                  unsigned short* __restrict__ xbf) {
    size_t i = ((size_t)blockIdx.x * blockDim.x + threadIdx.x) * 4;
    float4 v = *(const float4*)(x + i);
    ushort4 o;
    o.x = f2bf_rne(v.x); o.y = f2bf_rne(v.y);
    o.z = f2bf_rne(v.z); o.w = f2bf_rne(v.w);
    *(ushort4*)(xbf + i) = o;
}

// ---------------------------------------------------------------------------
// Kernel 2: embed fp32 -> bf16, plus per-row sum of squares (e_sq).
// One 256-thread block per codebook row.
// ---------------------------------------------------------------------------
__global__ void vq_convert_embed(const float* __restrict__ e,
                                 unsigned short* __restrict__ ebf,
                                 float* __restrict__ e_sq) {
    __shared__ float part[8];
    int row = blockIdx.x;
    int t = threadIdx.x;
    float v = e[(size_t)row * DIM + t];
    ebf[(size_t)row * DIM + t] = f2bf_rne(v);
    float s = v * v;
    #pragma unroll
    for (int off = 16; off >= 1; off >>= 1)
        s += __shfl_xor(s, off, 32);
    if ((t & 31) == 0) part[t >> 5] = s;
    __syncthreads();
    if (t == 0) {
        float tot = 0.f;
        #pragma unroll
        for (int w = 0; w < 8; ++w) tot += part[w];
        e_sq[row] = tot;
    }
}

// ---------------------------------------------------------------------------
// Kernel 3: fused GEMM + running argmin.
//  Block = 256 threads = 8 waves; each wave owns 16 rows (A resident in VGPRs).
//  Per iteration the block stages 64 codebook rows into LDS (padded stride),
//  every wave runs 4 column tiles x 8 WMMA(16x16x32 bf16) over K=256.
// ---------------------------------------------------------------------------
#define NT 64                 // codes staged per iteration
#define B_STRIDE 264          // halves per staged row (256 + 8 pad -> no bank clash)

__global__ void __launch_bounds__(256)
vq_argmin_wmma(const unsigned short* __restrict__ Abf,
               const unsigned short* __restrict__ Ebf,
               const float* __restrict__ e_sq,
               int* __restrict__ idx_out,
               float* __restrict__ idx_out_f) {
    __shared__ unsigned short Bt[NT * B_STRIDE];
    __shared__ float esq_s[NT];

    const int tid    = threadIdx.x;
    const int lane   = tid & 31;
    const int laneLo = lane & 15;
    const int hi     = lane >> 4;            // lane half: 0 or 1
    const int waveId = tid >> 5;
    const int rowBase = blockIdx.x * 128 + waveId * 16;

    // ---- Load resident A tile: 16 rows x 256 halves, ISA 16-bit A layout.
    // chunk c covers K = [32c, 32c+32): halves 0..7 at K = 32c + hi*8,
    // halves 8..15 at K = 32c + hi*8 + 16.
    Frag16 a[8];
    {
        const unsigned short* arow = Abf + (size_t)(rowBase + laneLo) * DIM;
        #pragma unroll
        for (int c = 0; c < 8; ++c) {
            const unsigned short* ap = arow + 32 * c + hi * 8;
            a[c].q[0] = *(const uint4*)ap;
            a[c].q[1] = *(const uint4*)(ap + 16);
        }
    }

    float bestD[8];
    int   bestI[8];
    #pragma unroll
    for (int r = 0; r < 8; ++r) { bestD[r] = 3.4e38f; bestI[r] = 0; }

    const int NITER = N_CODES / NT;          // 128
    for (int it = 0; it < NITER; ++it) {
        const int nbase = it * NT;

        // ---- Stage 64 codebook rows (bf16) into LDS: 4 threads/row, 64B each.
        {
            const int j = tid >> 2;          // staged row 0..63
            const int p = tid & 3;
            const uint4* src = (const uint4*)(Ebf + (size_t)(nbase + j) * DIM);
            uint4* dst = (uint4*)(&Bt[j * B_STRIDE]);
            #pragma unroll
            for (int q = 0; q < 4; ++q)
                dst[p * 4 + q] = src[p * 4 + q];
            if (tid < NT) esq_s[tid] = e_sq[nbase + tid];
        }
        __syncthreads();

        // ---- Prefetch next tile into cache while we compute this one.
        if (it + 1 < NITER) {
            const char* pf = (const char*)(Ebf + (size_t)(nbase + NT + (tid >> 2)) * DIM)
                             + (size_t)(tid & 3) * 128;
            __builtin_prefetch(pf, 0, 3);
        }

        // ---- 4 column tiles of 16 codes each.
        #pragma unroll
        for (int ct = 0; ct < 4; ++ct) {
            v8f acc = {};
            const unsigned short* brow = &Bt[(ct * 16 + laneLo) * B_STRIDE];
            #pragma unroll
            for (int c = 0; c < 8; ++c) {
                // B fragment (32x16 K x N): lane n = column, halves 0..7 at
                // K = 32c + hi*8, halves 8..15 at K = 32c + hi*8 + 16.
                Frag16 b;
                const unsigned short* bp = brow + 32 * c + hi * 8;
                b.q[0] = *(const uint4*)bp;
                b.q[1] = *(const uint4*)(bp + 16);
                acc = __builtin_amdgcn_wmma_f32_16x16x32_bf16(
                        false, a[c].v, false, b.v, (short)0, acc, false, false);
            }
            // dist proxy = e_sq[col] - 2 * score ; update running argmin.
            const float ecol = esq_s[ct * 16 + laneLo];
            const int   ncol = nbase + ct * 16 + laneLo;
            #pragma unroll
            for (int r = 0; r < 8; ++r) {
                float dist = ecol - 2.0f * acc[r];
                if (dist < bestD[r]) { bestD[r] = dist; bestI[r] = ncol; }
            }
        }
        __syncthreads();
    }

    // ---- Cross-lane argmin within each 16-lane group (C/D layout:
    // VGPR r -> row M=r for lanes 0-15, M=r+8 for lanes 16-31).
    #pragma unroll
    for (int r = 0; r < 8; ++r) {
        float d = bestD[r];
        int   i = bestI[r];
        #pragma unroll
        for (int off = 1; off < 16; off <<= 1) {
            float od = __shfl_xor(d, off, 32);
            int   oi = __shfl_xor(i, off, 32);
            if (od < d || (od == d && oi < i)) { d = od; i = oi; }
        }
        if (laneLo == 0) {
            int row = rowBase + r + hi * 8;
            idx_out[row]   = i;
            idx_out_f[row] = (float)i;
        }
    }
}

// ---------------------------------------------------------------------------
// Kernel 4: quantize = embed[idx] (exact fp32 gather).
// 256 threads = 4 rows/block, 64 threads copy one row as float4s.
// ---------------------------------------------------------------------------
__global__ void vq_gather(const float* __restrict__ embed,
                          const int* __restrict__ idx,
                          float* __restrict__ out) {
    int row = blockIdx.x * 4 + (threadIdx.x >> 6);
    int l   = threadIdx.x & 63;
    int k   = idx[row];
    const float4* src = (const float4*)(embed + (size_t)k * DIM);
    float4* dst = (float4*)(out + (size_t)row * DIM);
    dst[l] = src[l];
}

// ---------------------------------------------------------------------------
extern "C" void kernel_launch(void* const* d_in, const int* in_sizes, int n_in,
                              void* d_out, int out_size, void* d_ws, size_t ws_size,
                              hipStream_t stream) {
    const float* x = (const float*)d_in[0];     // [32768, 256] fp32
    const float* e = (const float*)d_in[1];     // [8192, 256]  fp32

    float* quant   = (float*)d_out;                         // N*DIM floats
    float* idx_f   = (float*)d_out + (size_t)N_ROWS * DIM;  // N floats (indices)

    // Workspace layout (bytes):
    char* ws = (char*)d_ws;
    unsigned short* xbf  = (unsigned short*)(ws);                        // 16 MB
    unsigned short* ebf  = (unsigned short*)(ws + (size_t)N_ROWS * DIM * 2);        // 4 MB
    float*          e_sq = (float*)(ws + (size_t)N_ROWS * DIM * 2
                                       + (size_t)N_CODES * DIM * 2);     // 32 KB
    int*            idxi = (int*)((char*)e_sq + (size_t)N_CODES * 4);    // 128 KB

    // 1) X -> bf16
    vq_convert_inputs<<<(N_ROWS * DIM) / (256 * 4), 256, 0, stream>>>(x, xbf);
    // 2) E -> bf16 + row norms
    vq_convert_embed<<<N_CODES, 256, 0, stream>>>(e, ebf, e_sq);
    // 3) fused WMMA GEMM + argmin
    vq_argmin_wmma<<<N_ROWS / 128, 256, 0, stream>>>(xbf, ebf, e_sq, idxi, idx_f);
    // 4) gather codebook rows (fp32 exact)
    vq_gather<<<N_ROWS / 4, 256, 0, stream>>>(e, idxi, quant);
}
